// SimpleRetopologyModel_42992622632991
// MI455X (gfx1250) — matrix operand order repaired and checked by reading, not compiled, not run on hardware
//
#include <hip/hip_runtime.h>
#include <hip/hip_bf16.h>

typedef __attribute__((ext_vector_type(16))) _Float16 v16h;
typedef __attribute__((ext_vector_type(8)))  float    v8f;

// ---------------------------------------------------------------------------
// Elementwise / scatter kernels
// ---------------------------------------------------------------------------

__global__ void zero_f32_kernel(float* __restrict__ p, int n) {
    int i = blockIdx.x * blockDim.x + threadIdx.x;
    int stride = gridDim.x * blockDim.x;
    for (; i < n; i += stride) p[i] = 0.0f;
}

__global__ void degree_kernel(const int* __restrict__ dst, int nE,
                              float* __restrict__ deg) {
    int i = blockIdx.x * blockDim.x + threadIdx.x;
    if (i < nE) atomicAdd(&deg[dst[i]], 1.0f);
}

__global__ void dinv_kernel(float* __restrict__ deg_dinv, int n) {
    int i = blockIdx.x * blockDim.x + threadIdx.x;
    if (i < n) deg_dinv[i] = rsqrtf(deg_dinv[i] + 1.0f);
}

// y1[node,c] = (x[node,0:6] @ W1[:,c]) * dinv[node]   (K=6: scalar FMA is optimal)
__global__ void y1_kernel(const float* __restrict__ x, const float* __restrict__ W1,
                          const float* __restrict__ dinv, float* __restrict__ y, int n) {
    int idx = blockIdx.x * blockDim.x + threadIdx.x;
    if (idx >= n * 32) return;
    int node = idx >> 5, c = idx & 31;
    const float* xr = x + (long)node * 6;
    float acc = 0.0f;
#pragma unroll
    for (int k = 0; k < 6; ++k) acc += xr[k] * W1[k * 32 + c];
    y[idx] = acc * dinv[node];
}

// acc[dst, :] += y[src, :]  — 8 threads/edge, float4 gathers, f32 atomics (resolve in L2).
// Prefetch the (sequential) edge-index stream so the atomic pipe stays fed.
__global__ void scatter_kernel(const int* __restrict__ src, const int* __restrict__ dst,
                               const float* __restrict__ y, float* __restrict__ acc, int nE) {
    int idx = blockIdx.x * blockDim.x + threadIdx.x;
    if (idx >= nE * 8) return;
    int e = idx >> 3, q = idx & 7;
    __builtin_prefetch(src + e + 2048, 0, 1);   // global_prefetch_b8 (speculative, OOB dropped)
    __builtin_prefetch(dst + e + 2048, 0, 1);
    int s = src[e], d = dst[e];
    const float4 v = *(const float4*)(y + (long)s * 32 + q * 4);
    float* a = acc + (long)d * 32 + q * 4;
    atomicAdd(a + 0, v.x);
    atomicAdd(a + 1, v.y);
    atomicAdd(a + 2, v.z);
    atomicAdd(a + 3, v.w);
}

// h = relu(dinv*(acc + y) + b1); re-zero acc for layer 2
__global__ void h_kernel(const float* __restrict__ dinv, const float* __restrict__ y,
                         float* __restrict__ acc, const float* __restrict__ b1,
                         float* __restrict__ h, int n) {
    int idx = blockIdx.x * blockDim.x + threadIdx.x;
    if (idx >= n * 32) return;
    int node = idx >> 5, c = idx & 31;
    float v = dinv[node] * (acc[idx] + y[idx]) + b1[c];
    h[idx] = fmaxf(v, 0.0f);
    acc[idx] = 0.0f;
}

// ---------------------------------------------------------------------------
// WMMA kernels — one 16-node tile per 32-lane wave (single-wave blocks so
// EXEC is all-1s, as the WMMA ISA requires)
// ---------------------------------------------------------------------------

// A-matrix (16x32, f16) element->K map: contiguous K = g*16+e (matches B fill)
__device__ inline int a_k_index(int g, int e) {
    return g * 16 + e;
}

// y2 = (h @ W2) * dinv[row]  via 2x v_wmma_f32_16x16x32_f16.
// Results staged in LDS, written back as coalesced float4 (global_store_b128).
__launch_bounds__(32)
__global__ void y2_wmma_kernel(const float* __restrict__ h, const float* __restrict__ W2,
                               const float* __restrict__ dinv, float* __restrict__ y, int n) {
    __shared__ float tile[16 * 32];
    const int lane = threadIdx.x & 31;
    const int l15  = lane & 15;          // A: row m ; B/C: column n
    const int g    = lane >> 4;
    const int base = blockIdx.x * 16;
    const bool full = (base + 16) <= n;  // wave-uniform

    int row = base + l15; if (row >= n) row = n - 1;
    const float* hr = h + (long)row * 32;
    v16h a;
#pragma unroll
    for (int e = 0; e < 16; ++e) a[e] = (_Float16)hr[a_k_index(g, e)];

    v16h b_lo, b_hi;                     // W2[:,0:16] and W2[:,16:32], B layout K=g*16+e
#pragma unroll
    for (int e = 0; e < 16; ++e) {
        int k = g * 16 + e;
        b_lo[e] = (_Float16)W2[k * 32 + l15];
        b_hi[e] = (_Float16)W2[k * 32 + 16 + l15];
    }

    v8f c0 = {}, c1 = {};
    c0 = __builtin_amdgcn_wmma_f32_16x16x32_f16(false, a, false, b_lo, (short)0, c0, false, false);
    c1 = __builtin_amdgcn_wmma_f32_16x16x32_f16(false, a, false, b_hi, (short)0, c1, false, false);

    // C layout: element r -> row (r + 8g), column l15 / l15+16. Scale rows, stage in LDS.
#pragma unroll
    for (int r = 0; r < 8; ++r) {
        int m  = r + 8 * g;
        int gm = base + m; if (gm >= n) gm = n - 1;
        float dv = dinv[gm];
        tile[m * 32 + l15]      = c0[r] * dv;
        tile[m * 32 + 16 + l15] = c1[r] * dv;
    }
    __syncthreads();
    // Coalesced write-back: 512 floats = 32 lanes x 4 float4
    if (full) {
#pragma unroll
        for (int t = 0; t < 4; ++t) {
            int off = t * 128 + lane * 4;   // element offset within tile
            *(float4*)(y + (long)base * 32 + off) = *(const float4*)(tile + off);
        }
    } else {
#pragma unroll
        for (int t = 0; t < 4; ++t) {
            int off = t * 128 + lane * 4;
            if (base + (off >> 5) < n)
                *(float4*)(y + (long)base * 32 + off) = *(const float4*)(tile + off);
        }
    }
}

// features = dinv*(acc + y) + b2  (stored to out) ; then both MLP heads.
// Head hidden layers (32->16) via WMMA; 16->1 / 16->6 tails via LDS reduction.
__launch_bounds__(32)
__global__ void feat_heads_kernel(const float* __restrict__ dinv, const float* __restrict__ y,
                                  const float* __restrict__ acc, const float* __restrict__ b2,
                                  const float* __restrict__ Wt1, const float* __restrict__ bt1,
                                  const float* __restrict__ Wt2, const float* __restrict__ bt2,
                                  const float* __restrict__ Wa1, const float* __restrict__ ba1,
                                  const float* __restrict__ Wa2, const float* __restrict__ ba2,
                                  float* __restrict__ out, int n) {
    __shared__ float ftile[16 * 32];     // feature tile (for coalesced store)
    __shared__ float sh[16 * 17];        // 16x16 head-hidden tile, padded stride
    const int lane = threadIdx.x & 31;
    const int l15  = lane & 15;
    const int g    = lane >> 4;
    const int base = blockIdx.x * 16;
    const bool full = (base + 16) <= n;  // wave-uniform
    float* f_out = out + (long)7 * n;    // features region of d_out

    // Build features tile directly in WMMA A layout (each (m,k) produced once)
    int row = base + l15;
    int rr  = (row < n) ? row : (n - 1);
    float dv = dinv[rr];
    v16h a;
#pragma unroll
    for (int e = 0; e < 16; ++e) {
        int k = a_k_index(g, e);
        float fv = dv * (acc[(long)rr * 32 + k] + y[(long)rr * 32 + k]) + b2[k];
        ftile[l15 * 32 + k] = fv;
        a[e] = (_Float16)fv;
    }
    __syncthreads();
    // Coalesced feature store (float4 lines)
    if (full) {
#pragma unroll
        for (int t = 0; t < 4; ++t) {
            int off = t * 128 + lane * 4;
            *(float4*)(f_out + (long)base * 32 + off) = *(const float4*)(ftile + off);
        }
    } else {
#pragma unroll
        for (int t = 0; t < 4; ++t) {
            int off = t * 128 + lane * 4;
            if (base + (off >> 5) < n)
                *(float4*)(f_out + (long)base * 32 + off) = *(const float4*)(ftile + off);
        }
    }

    // --- topology head: relu(f @ Wt1 + bt1) @ Wt2 + bt2 ---
    v16h bmat;
#pragma unroll
    for (int e = 0; e < 16; ++e) bmat[e] = (_Float16)Wt1[(g * 16 + e) * 16 + l15];
    v8f ct = {};
    ct = __builtin_amdgcn_wmma_f32_16x16x32_f16(false, a, false, bmat, (short)0, ct, false, false);
#pragma unroll
    for (int r = 0; r < 8; ++r)
        sh[(r + 8 * g) * 17 + l15] = fmaxf(ct[r] + bt1[l15], 0.0f);
    __syncthreads();
    if (lane < 16) {
        int m = base + lane;
        if (m < n) {
            float t = bt2[0];
#pragma unroll
            for (int j = 0; j < 16; ++j) t += sh[lane * 17 + j] * Wt2[j];
            out[m] = t;                  // topology_logits
        }
    }
    __syncthreads();

    // --- attribute head: relu(f @ Wa1 + ba1) @ Wa2 + ba2 ---
#pragma unroll
    for (int e = 0; e < 16; ++e) bmat[e] = (_Float16)Wa1[(g * 16 + e) * 16 + l15];
    v8f ca = {};
    ca = __builtin_amdgcn_wmma_f32_16x16x32_f16(false, a, false, bmat, (short)0, ca, false, false);
#pragma unroll
    for (int r = 0; r < 8; ++r)
        sh[(r + 8 * g) * 17 + l15] = fmaxf(ca[r] + ba1[l15], 0.0f);
    __syncthreads();
    if (lane < 16) {
        int m = base + lane;
        if (m < n) {
#pragma unroll
            for (int o = 0; o < 6; ++o) {
                float v = ba2[o];
#pragma unroll
                for (int j = 0; j < 16; ++j) v += sh[lane * 17 + j] * Wa2[j * 6 + o];
                if (o < 3) out[(long)n + (long)m * 3 + o]            = v;   // normals
                else       out[(long)4 * n + (long)m * 3 + (o - 3)]  = v;   // uvs
            }
        }
    }
}

// ---------------------------------------------------------------------------
// Launch
// ---------------------------------------------------------------------------

extern "C" void kernel_launch(void* const* d_in, const int* in_sizes, int n_in,
                              void* d_out, int out_size, void* d_ws, size_t ws_size,
                              hipStream_t stream) {
    const float* x    = (const float*)d_in[0];
    const int*   edge = (const int*)  d_in[1];
    const float* W1   = (const float*)d_in[2];
    const float* b1   = (const float*)d_in[3];
    const float* W2   = (const float*)d_in[4];
    const float* b2   = (const float*)d_in[5];
    const float* Wt1  = (const float*)d_in[6];
    const float* bt1  = (const float*)d_in[7];
    const float* Wt2  = (const float*)d_in[8];
    const float* bt2  = (const float*)d_in[9];
    const float* Wa1  = (const float*)d_in[10];
    const float* ba1  = (const float*)d_in[11];
    const float* Wa2  = (const float*)d_in[12];
    const float* ba2  = (const float*)d_in[13];

    const int n  = in_sizes[0] / 6;
    const int nE = in_sizes[1] / 2;
    const int* src = edge;
    const int* dst = edge + nE;

    float* ws  = (float*)d_ws;
    float* deg = ws;                 // n floats (becomes dinv)
    float* acc = ws + n;             // 32n
    float* y   = ws + (long)33 * n;  // 32n (y1 then y2)
    float* h   = ws + (long)65 * n;  // 32n
    float* out = (float*)d_out;

    const int T = 256;
    const int tiles = (n + 15) / 16;

    // zero deg + acc (contiguous 33n floats)
    {
        int zn = 33 * n;
        int blocks = (zn + T - 1) / T;
        if (blocks > 65535) blocks = 65535;
        zero_f32_kernel<<<blocks, T, 0, stream>>>(deg, zn);
    }
    degree_kernel<<<(nE + T - 1) / T, T, 0, stream>>>(dst, nE, deg);
    dinv_kernel<<<(n + T - 1) / T, T, 0, stream>>>(deg, n);

    // layer 1
    y1_kernel<<<(n * 32 + T - 1) / T, T, 0, stream>>>(x, W1, deg, y, n);
    scatter_kernel<<<(nE * 8 + T - 1) / T, T, 0, stream>>>(src, dst, y, acc, nE);
    h_kernel<<<(n * 32 + T - 1) / T, T, 0, stream>>>(deg, y, acc, b1, h, n);

    // layer 2 (WMMA) + scatter
    y2_wmma_kernel<<<tiles, 32, 0, stream>>>(h, W2, deg, y, n);
    scatter_kernel<<<(nE * 8 + T - 1) / T, T, 0, stream>>>(src, dst, y, acc, nE);

    // features + heads (WMMA)
    feat_heads_kernel<<<tiles, 32, 0, stream>>>(deg, y, acc, b2,
                                                Wt1, bt1, Wt2, bt2,
                                                Wa1, ba1, Wa2, ba2,
                                                out, n);
}
